// SparseGraphAttention_68453188764121
// MI455X (gfx1250) — matrix operand order repaired and checked by reading, not compiled, not run on hardware
//
#include <hip/hip_runtime.h>
#include <hip/hip_bf16.h>

#define DIM_H    128
#define N_HEADS  8
#define HEAD_DIM 16

typedef __attribute__((ext_vector_type(16))) _Float16 v16h;
typedef __attribute__((ext_vector_type(8)))  float    v8f;

// ---------------------------------------------------------------------------
// A-matrix 16x32 f16 fragment from f32 memory (ISA 7.12.2 layout).
// lane&15 = row. Lanes 0-15: VGPR0-3 hold K=0..7, VGPR4-7 hold K=16..23.
// Lanes 16-31: +8 on those K ranges.
// ---------------------------------------------------------------------------
__device__ __forceinline__ v16h load_a_frag(const float* __restrict__ A,
                                            int row0, int nrows, int kbase, int lane) {
    int row = row0 + (lane & 15);
    if (row >= nrows) row = nrows - 1;          // tail clamp (stores are guarded)
    const float* p = A + (size_t)row * DIM_H + kbase;
    int hi = (lane >> 4) * 8;
    v16h a;
#pragma unroll
    for (int v = 0; v < 8; ++v) {
        int k = ((v & 4) << 2) + hi + (v & 3) * 2;   // (v>=4?16:0) + hi + (v%4)*2
        a[2 * v]     = (_Float16)p[k];
        a[2 * v + 1] = (_Float16)p[k + 1];
    }
    return a;
}

// ---------------------------------------------------------------------------
// Kernel 0: zero init (outp | denom | mkey are contiguous in workspace)
// ---------------------------------------------------------------------------
__global__ void sga_init_zero(float* __restrict__ p, int n) {
    int i = blockIdx.x * blockDim.x + threadIdx.x;
    if (i < n) p[i] = 0.0f;
}

// ---------------------------------------------------------------------------
// Kernel 0b: convert Wk/Wv/Wo (f32, row-major) into f16 B-fragments stored in
// exact per-lane WMMA layout. Fragment (ntile, kstep): lane&15 = column n,
// lanes 0-15 hold K=kstep*32+{0..7,16..23}, lanes 16-31 hold +8 (same as A).
// Buffer layout: [ntile*4 + kstep][lane][16 halfs]  (1024 B per fragment).
// One thread per packed dword: 8192 dwords per weight, 3 weights.
// ---------------------------------------------------------------------------
__global__ void sga_cvt_weights(const float* __restrict__ Wk,
                                const float* __restrict__ Wv,
                                const float* __restrict__ Wo,
                                _Float16* __restrict__ WkF,
                                _Float16* __restrict__ WvF,
                                _Float16* __restrict__ WoF) {
    int t = blockIdx.x * blockDim.x + threadIdx.x;
    const int DW_PER_W = (DIM_H * DIM_H) / 2;   // 8192
    if (t >= 3 * DW_PER_W) return;
    int w = t / DW_PER_W;
    int r = t - w * DW_PER_W;

    int tile = r >> 8;           // 0..31 = ntile*4 + kstep
    int lane = (r >> 3) & 31;
    int v    = r & 7;
    int ntile = tile >> 2, ks = tile & 3;
    int n = ntile * 16 + (lane & 15);
    int k = ks * 32 + ((v & 4) << 2) + (lane >> 4) * 8 + (v & 3) * 2;

    const float* W = (w == 0) ? Wk : (w == 1) ? Wv : Wo;
    _Float16*    O = (w == 0) ? WkF : (w == 1) ? WvF : WoF;
    O[(size_t)r * 2 + 0] = (_Float16)W[(size_t)n * DIM_H + k];
    O[(size_t)r * 2 + 1] = (_Float16)W[(size_t)n * DIM_H + k + 1];
}

// ---------------------------------------------------------------------------
// Kernel 1: fused K,V projection.  K = kv@Wk^T+bk,  V = kv@Wv^T+bv.
// block = 256 (8 waves = 8 column tiles), grid = ceil(N/16).
// A tile staged once in LDS (waves 0-3 convert one k-step each); B fragments
// come pre-swizzled f16 from global (two b128 loads, no cvt).
// ---------------------------------------------------------------------------
__global__ void sga_proj_kv(const float* __restrict__ kv,
                            const v16h* __restrict__ WkF, const float* __restrict__ bk,
                            const v16h* __restrict__ WvF, const float* __restrict__ bv,
                            float* __restrict__ Kout, float* __restrict__ Vout, int N) {
    __shared__ v16h aFrag[4][32];               // 4 KB
    const int lane = threadIdx.x & 31;
    const int wave = threadIdx.x >> 5;
    const int m0   = blockIdx.x * 16;
    if (m0 >= N) return;

    if (wave < 4)
        aFrag[wave][lane] = load_a_frag(kv, m0, N, wave * 32, lane);
    __syncthreads();

    const int n0 = wave * 16;                   // column tile
    v8f ck = {};
    v8f cv = {};
#pragma unroll
    for (int ks = 0; ks < 4; ++ks) {
        v16h a  = aFrag[ks][lane];              // ds_load_b128 x2
        v16h bK = WkF[(size_t)(wave * 4 + ks) * 32 + lane];
        v16h bV = WvF[(size_t)(wave * 4 + ks) * 32 + lane];
        ck = __builtin_amdgcn_wmma_f32_16x16x32_f16(false, a, false, bK,
                                                    (short)0, ck, false, false);
        cv = __builtin_amdgcn_wmma_f32_16x16x32_f16(false, a, false, bV,
                                                    (short)0, cv, false, false);
    }

    const int n  = n0 + (lane & 15);
    const int hi = (lane >> 4) * 8;
    const float biask = bk[n];
    const float biasv = bv[n];
#pragma unroll
    for (int r = 0; r < 8; ++r) {
        int m = m0 + r + hi;
        if (m < N) {
            Kout[(size_t)m * DIM_H + n] = ck[r] + biask;
            Vout[(size_t)m * DIM_H + n] = cv[r] + biasv;
        }
    }
}

// ---------------------------------------------------------------------------
// Kernel 2: per-(edge,head) scores + segment max via monotone-uint atomicMax.
// Threads t = e*8+h: 8 consecutive threads stream one full 512B node row.
// ---------------------------------------------------------------------------
__global__ void sga_edge_scores(const float* __restrict__ K, const float* __restrict__ q,
                                const int* __restrict__ src, const int* __restrict__ dst,
                                float* __restrict__ scores, unsigned* __restrict__ mkey,
                                int E) {
    int t = blockIdx.x * blockDim.x + threadIdx.x;
    if (t >= E * N_HEADS) return;
    int e = t >> 3, h = t & 7;
    int s = src[e], d = dst[e];

    const float4* kp = (const float4*)(K + (size_t)s * DIM_H + h * HEAD_DIM);
    const float4* qp = (const float4*)(q + (size_t)d * DIM_H + h * HEAD_DIM);
    float acc = 0.0f;
#pragma unroll
    for (int i = 0; i < 4; ++i) {
        float4 a = kp[i], b = qp[i];
        acc = fmaf(a.x, b.x, acc); acc = fmaf(a.y, b.y, acc);
        acc = fmaf(a.z, b.z, acc); acc = fmaf(a.w, b.w, acc);
    }
    acc *= 0.25f;                       // 1/sqrt(HEAD_DIM=16)
    scores[t] = acc;

    // order-preserving float -> uint key: neg -> ~bits, pos -> bits|0x80000000
    unsigned bits = __float_as_uint(acc);
    unsigned key  = (bits & 0x80000000u) ? ~bits : (bits | 0x80000000u);
    atomicMax(&mkey[(size_t)d * N_HEADS + h], key);   // GLOBAL_ATOMIC_MAX_U32
}

// ---------------------------------------------------------------------------
// Kernel 3: ex = exp(score - m[dst]) (in place) + denom scatter-add.
// ---------------------------------------------------------------------------
__global__ void sga_edge_exp(float* __restrict__ scores, const int* __restrict__ dst,
                             const unsigned* __restrict__ mkey,
                             float* __restrict__ denom, int E) {
    int t = blockIdx.x * blockDim.x + threadIdx.x;
    if (t >= E * N_HEADS) return;
    int e = t >> 3, h = t & 7;
    int d = dst[e];

    unsigned key  = mkey[(size_t)d * N_HEADS + h];
    unsigned bits = (key & 0x80000000u) ? (key & 0x7FFFFFFFu) : ~key;
    float m = __uint_as_float(bits);

    float v = __expf(scores[t] - m);
    scores[t] = v;
    atomicAdd(&denom[(size_t)d * N_HEADS + h], v);    // GLOBAL_ATOMIC_ADD_F32
}

// ---------------------------------------------------------------------------
// Kernel 4: weighted scatter-add:  outp[dst] += V[src] * ex/denom[dst]
// ---------------------------------------------------------------------------
__global__ void sga_edge_aggregate(const float* __restrict__ V, const float* __restrict__ ex,
                                   const float* __restrict__ denom,
                                   const int* __restrict__ src, const int* __restrict__ dst,
                                   float* __restrict__ outp, int E) {
    int t = blockIdx.x * blockDim.x + threadIdx.x;
    if (t >= E * N_HEADS) return;
    int e = t >> 3, h = t & 7;
    int s = src[e], d = dst[e];

    float a = ex[t] / (denom[(size_t)d * N_HEADS + h] + 1e-16f);
    const float4* vp = (const float4*)(V + (size_t)s * DIM_H + h * HEAD_DIM);
    float* op = outp + (size_t)d * DIM_H + h * HEAD_DIM;
#pragma unroll
    for (int i = 0; i < 4; ++i) {
        float4 v = vp[i];
        atomicAdd(&op[4 * i + 0], v.x * a);
        atomicAdd(&op[4 * i + 1], v.y * a);
        atomicAdd(&op[4 * i + 2], v.z * a);
        atomicAdd(&op[4 * i + 3], v.w * a);
    }
}

// ---------------------------------------------------------------------------
// Kernel 5: output projection  out = outp @ Wo^T + bo  (WMMA, LDS-staged A)
// ---------------------------------------------------------------------------
__global__ void sga_proj_out(const float* __restrict__ X,
                             const v16h* __restrict__ WoF, const float* __restrict__ bo,
                             float* __restrict__ out, int N) {
    __shared__ v16h aFrag[4][32];
    const int lane = threadIdx.x & 31;
    const int wave = threadIdx.x >> 5;
    const int m0   = blockIdx.x * 16;
    if (m0 >= N) return;

    if (wave < 4)
        aFrag[wave][lane] = load_a_frag(X, m0, N, wave * 32, lane);
    __syncthreads();

    const int n0 = wave * 16;
    v8f c = {};
#pragma unroll
    for (int ks = 0; ks < 4; ++ks) {
        v16h a = aFrag[ks][lane];
        v16h b = WoF[(size_t)(wave * 4 + ks) * 32 + lane];
        c = __builtin_amdgcn_wmma_f32_16x16x32_f16(false, a, false, b,
                                                   (short)0, c, false, false);
    }

    const int n  = n0 + (lane & 15);
    const int hi = (lane >> 4) * 8;
    const float bias = bo[n];
#pragma unroll
    for (int r = 0; r < 8; ++r) {
        int m = m0 + r + hi;
        if (m < N) out[(size_t)m * DIM_H + n] = c[r] + bias;
    }
}

// ---------------------------------------------------------------------------
// Launch
// ---------------------------------------------------------------------------
extern "C" void kernel_launch(void* const* d_in, const int* in_sizes, int n_in,
                              void* d_out, int out_size, void* d_ws, size_t ws_size,
                              hipStream_t stream) {
    const float* q  = (const float*)d_in[0];
    const float* kv = (const float*)d_in[1];
    const int*   ei = (const int*)  d_in[2];
    const float* Wk = (const float*)d_in[3];
    const float* bk = (const float*)d_in[4];
    const float* Wv = (const float*)d_in[5];
    const float* bv = (const float*)d_in[6];
    const float* Wo = (const float*)d_in[7];
    const float* bo = (const float*)d_in[8];

    const int N = in_sizes[0] / DIM_H;
    const int E = in_sizes[2] / 2;
    const int* src = ei;          // edge_index[0]
    const int* dst = ei + E;      // edge_index[1]

    // Workspace layout:
    //   Kf[N*128] | Vf[N*128] | ex[E*8] | outp[N*128] | denom[N*8] | mkey[N*8]
    //   | WkF[128*128 f16] | WvF[...] | WoF[...]
    float*    Kf    = (float*)d_ws;
    float*    Vf    = Kf   + (size_t)N * DIM_H;
    float*    ex    = Vf   + (size_t)N * DIM_H;
    float*    outp  = ex   + (size_t)E * N_HEADS;
    float*    denom = outp + (size_t)N * DIM_H;
    unsigned* mkey  = (unsigned*)(denom + (size_t)N * N_HEADS);
    _Float16* WkF   = (_Float16*)(mkey + (size_t)N * N_HEADS);
    _Float16* WvF   = WkF + (size_t)DIM_H * DIM_H;
    _Float16* WoF   = WvF + (size_t)DIM_H * DIM_H;

    const int mtiles   = (N + 15) / 16;
    const int eThreads = E * N_HEADS;
    const int eBlocks  = (eThreads + 255) / 256;

    // 0) zero outp + denom + mkey (contiguous region)
    const int initN = N * DIM_H + 2 * N * N_HEADS;
    sga_init_zero<<<(initN + 255) / 256, 256, 0, stream>>>(outp, initN);

    // 0b) pre-swizzle all three weight matrices into f16 WMMA B-fragments
    const int cvtN = 3 * (DIM_H * DIM_H) / 2;
    sga_cvt_weights<<<(cvtN + 255) / 256, 256, 0, stream>>>(Wk, Wv, Wo, WkF, WvF, WoF);

    // 1) K,V projections (WMMA, LDS-staged A, f16 B fragments)
    sga_proj_kv<<<mtiles, 256, 0, stream>>>(kv, (const v16h*)WkF, bk,
                                            (const v16h*)WvF, bv, Kf, Vf, N);

    // 2) edge scores + segment max
    sga_edge_scores<<<eBlocks, 256, 0, stream>>>(Kf, q, src, dst, ex, mkey, E);

    // 3) exp + segment sum
    sga_edge_exp<<<eBlocks, 256, 0, stream>>>(ex, dst, mkey, denom, E);

    // 4) weighted scatter-add aggregation
    sga_edge_aggregate<<<eBlocks, 256, 0, stream>>>(Vf, ex, denom, src, dst, outp, E);

    // 5) output projection (WMMA) -> d_out
    sga_proj_out<<<mtiles, 256, 0, stream>>>(outp, (const v16h*)WoF, bo,
                                             (float*)d_out, N);
}
